// AttentionDecoder_26980984553623
// MI455X (gfx1250) — matrix operand order repaired and checked by reading, not compiled, not run on hardware
//
#include <hip/hip_runtime.h>
#include <hip/hip_bf16.h>
#include <math.h>

// ---------------------------------------------------------------------------
// Problem constants (from reference)
// ---------------------------------------------------------------------------
namespace {
constexpr int B = 256, S = 128, T_STEPS = 16;
constexpr int D = 512, H = 512, E = 128, ADIM = 256, V = 37;
constexpr int V_PAD = 64;       // logits N padded to tile multiple (zero rows)
constexpr int SOS = 36;
constexpr int GH = 4 * H;       // 2048 gate width
constexpr int KX = E + D + H;   // 1152: [emb | ctx | h] concat K for gates GEMM
constexpr int KY = H + D;       // 1024: [h | ctx] concat K for logits GEMM

typedef __attribute__((ext_vector_type(4)))  unsigned short ushort4v;
typedef __attribute__((ext_vector_type(8)))  unsigned short ushort8;
typedef __attribute__((ext_vector_type(16))) __bf16        v16bf;
typedef __attribute__((ext_vector_type(8)))  float         v8f;

union Frag { ushort8 u8[2]; v16bf bf; };

__device__ __forceinline__ unsigned short f2bfu(float x) {
  // f32 -> bf16 with round-to-nearest-even
  unsigned u = __builtin_bit_cast(unsigned, x);
  u += 0x7FFFu + ((u >> 16) & 1u);
  return (unsigned short)(u >> 16);
}

__device__ __forceinline__ float sigf(float x) { return 1.0f / (1.0f + expf(-x)); }

// ---------------------------------------------------------------------------
// WMMA GEMM:  C[M,Ns] = act( A[M,K](f32) * W[Np,K](bf16)^T + bias[Ns] )
// Caller guarantees: M % 128 == 0, K % 64 == 0, Np % 64 == 0.
// Block: 256 threads = 8 waves; block tile 128(M) x 64(N); K-step 64;
// LDS double-buffered (one barrier per K-step, staging overlaps WMMA).
// Wave (wm,wn) = (wave>>1, wave&1) computes a 2x2 grid of 16x16 tiles:
// 8 x v_wmma_f32_16x16x32_bf16 per K-step, A/B fragments each reused twice.
// ---------------------------------------------------------------------------
constexpr int TM = 128, TN = 64, TK = 64;
constexpr int LDS_P = 72;  // row pitch (halves): 144B rows -> 16B-aligned b128 frags

__global__ __launch_bounds__(256) void gemm_bf16_wmma(
    const float* __restrict__ A, int lda,
    const unsigned short* __restrict__ Wb, int ldw,
    const float* __restrict__ bias,
    float* __restrict__ C, int ldc,
    int M, int Ns, int K, int act) {
  __shared__ __align__(16) unsigned short As[2][TM * LDS_P];  // 2 x 18 KB
  __shared__ __align__(16) unsigned short Ws[2][TN * LDS_P];  // 2 x 9 KB

  const int tid = threadIdx.x;
  const int m0 = blockIdx.x * TM;
  const int n0 = blockIdx.y * TN;
  const int wave = tid >> 5, lane = tid & 31;
  const int wm = wave >> 1, wn = wave & 1;   // 4 x 2 wave grid
  const int h = lane >> 4, r = lane & 15;    // half-wave select / index-in-half

  v8f acc[2][2] = {};

  // Stage one K-panel into LDS buffer bi. Loads are hoisted into registers
  // first so they clause and take a single loadcnt wait.
  auto stage = [&](int bi, int kof) {
    float4 va[8];
#pragma unroll
    for (int q = 0; q < 8; ++q) {
      int qq = tid + q * 256;
      int rr = qq >> 4, c4 = (qq & 15) << 2;
      va[q] = *reinterpret_cast<const float4*>(
          &A[(size_t)(m0 + rr) * lda + kof + c4]);
    }
    ushort8 wv[2];
#pragma unroll
    for (int q = 0; q < 2; ++q) {
      int qq = tid + q * 256;
      int rr = qq >> 3, c8 = (qq & 7) << 3;
      wv[q] = *reinterpret_cast<const ushort8*>(
          &Wb[(size_t)(n0 + rr) * ldw + kof + c8]);
    }
#pragma unroll
    for (int q = 0; q < 8; ++q) {
      int qq = tid + q * 256;
      int rr = qq >> 4, c4 = (qq & 15) << 2;
      ushort4v pv;
      pv[0] = f2bfu(va[q].x); pv[1] = f2bfu(va[q].y);
      pv[2] = f2bfu(va[q].z); pv[3] = f2bfu(va[q].w);
      *reinterpret_cast<ushort4v*>(&As[bi][rr * LDS_P + c4]) = pv;  // b64
    }
#pragma unroll
    for (int q = 0; q < 2; ++q) {
      int qq = tid + q * 256;
      int rr = qq >> 3, c8 = (qq & 7) << 3;
      *reinterpret_cast<ushort8*>(&Ws[bi][rr * LDS_P + c8]) = wv[q];  // b128
    }
  };

  auto compute = [&](int bi) {
#pragma unroll
    for (int kk = 0; kk < TK; kk += 32) {
      // A fragments 16x32 (ISA 7.12.2): half h -> K kk+h*8.. / kk+16+h*8..
      const unsigned short* ar0 = &As[bi][(wm * 32 + r) * LDS_P + kk];
      const unsigned short* ar1 = ar0 + 16 * LDS_P;
      Frag a0, a1;
      a0.u8[0] = *reinterpret_cast<const ushort8*>(ar0 + h * 8);
      a0.u8[1] = *reinterpret_cast<const ushort8*>(ar0 + 16 + h * 8);
      a1.u8[0] = *reinterpret_cast<const ushort8*>(ar1 + h * 8);
      a1.u8[1] = *reinterpret_cast<const ushort8*>(ar1 + 16 + h * 8);
      // B fragments 32x16: lane -> column N=r, half h -> K kk+h*8.. / kk+16+h*8..
      const unsigned short* wr0 = &Ws[bi][(wn * 32 + r) * LDS_P + kk + h * 8];
      const unsigned short* wr1 = wr0 + 16 * LDS_P;
      Frag b0, b1;
      b0.u8[0] = *reinterpret_cast<const ushort8*>(wr0);
      b0.u8[1] = *reinterpret_cast<const ushort8*>(wr0 + 16);
      b1.u8[0] = *reinterpret_cast<const ushort8*>(wr1);
      b1.u8[1] = *reinterpret_cast<const ushort8*>(wr1 + 16);

      acc[0][0] = __builtin_amdgcn_wmma_f32_16x16x32_bf16(
          false, a0.bf, false, b0.bf, (short)0, acc[0][0], false, false);
      acc[0][1] = __builtin_amdgcn_wmma_f32_16x16x32_bf16(
          false, a0.bf, false, b1.bf, (short)0, acc[0][1], false, false);
      acc[1][0] = __builtin_amdgcn_wmma_f32_16x16x32_bf16(
          false, a1.bf, false, b0.bf, (short)0, acc[1][0], false, false);
      acc[1][1] = __builtin_amdgcn_wmma_f32_16x16x32_bf16(
          false, a1.bf, false, b1.bf, (short)0, acc[1][1], false, false);
    }
  };

  const int nk = K / TK;
  stage(0, 0);
  __syncthreads();
  for (int ki = 0; ki < nk; ++ki) {
    const int cur = ki & 1;
    if (ki + 1 < nk) stage(cur ^ 1, (ki + 1) * TK);   // overlap with compute
    if (ki + 2 < nk)                                   // global_prefetch_b8
      __builtin_prefetch(&A[(size_t)(m0 + (tid & 127)) * lda + (ki + 2) * TK],
                         0, 1);
    compute(cur);
    __syncthreads();
  }

  // ---- Store: C/D layout: VGPR j -> M = h*8 + j, N = r (per 16x16 tile) ----
#pragma unroll
  for (int mt = 0; mt < 2; ++mt) {
    const int mbase = m0 + wm * 32 + mt * 16 + h * 8;
#pragma unroll
    for (int nt = 0; nt < 2; ++nt) {
      const int n = n0 + wn * 32 + nt * 16 + r;
      if (n < Ns) {
        const float bv = bias[n];
#pragma unroll
        for (int j = 0; j < 8; ++j) {
          float v = acc[mt][nt][j] + bv;
          if (act == 1) v = tanhf(v);
          C[(size_t)(mbase + j) * ldc + n] = v;
        }
      }
    }
  }
}

// ---------------------------------------------------------------------------
// Small helper kernels
// ---------------------------------------------------------------------------
__global__ void cvt_rows_kernel(unsigned short* __restrict__ dst, int ldd, int off,
                                const float* __restrict__ src, int rows, int cols) {
  int idx = blockIdx.x * blockDim.x + threadIdx.x;
  if (idx >= rows * cols) return;
  int rr = idx / cols, cc = idx % cols;
  dst[(size_t)rr * ldd + off + cc] = f2bfu(src[(size_t)rr * cols + cc]);
}

__global__ void fill0_u16_kernel(unsigned short* __restrict__ p, int n) {
  int idx = blockIdx.x * blockDim.x + threadIdx.x;
  if (idx < n) p[idx] = 0;
}

__global__ void bias_comb_kernel(float* __restrict__ bc,
                                 const float* __restrict__ b_ih,
                                 const float* __restrict__ b_hh) {
  int idx = blockIdx.x * blockDim.x + threadIdx.x;
  if (idx < GH) bc[idx] = b_ih[idx] + b_hh[idx];
}

__global__ void enc_mean_kernel(const float* __restrict__ enc, float* __restrict__ mean) {
  int idx = blockIdx.x * blockDim.x + threadIdx.x;
  if (idx >= B * D) return;
  int b = idx / D, d = idx % D;
  float s = 0.0f;
  for (int ss = 0; ss < S; ++ss) s += enc[((size_t)b * S + ss) * D + d];
  mean[idx] = s * (1.0f / (float)S);
}

__global__ void fill0_kernel(float* __restrict__ p, int n) {
  int idx = blockIdx.x * blockDim.x + threadIdx.x;
  if (idx < n) p[idx] = 0.0f;
}

__global__ void build_x_kernel(float* __restrict__ X, const float* __restrict__ emb,
                               const int* __restrict__ tgt, const float* __restrict__ ctx,
                               const float* __restrict__ h, int t) {
  int idx = blockIdx.x * blockDim.x + threadIdx.x;
  if (idx >= B * KX) return;
  int b = idx / KX, k = idx % KX;
  float v;
  if (k < E) {
    int tok = (t == 0) ? SOS : tgt[b * T_STEPS + (t - 1)];
    v = emb[(size_t)tok * E + k];
  } else if (k < E + D) {
    v = ctx[(size_t)b * D + (k - E)];
  } else {
    v = h[(size_t)b * H + (k - E - D)];
  }
  X[idx] = v;
}

__global__ void lstm_kernel(const float* __restrict__ gates,
                            float* __restrict__ h, float* __restrict__ c) {
  int idx = blockIdx.x * blockDim.x + threadIdx.x;
  if (idx >= B * H) return;
  int b = idx / H, j = idx % H;
  const float* g = gates + (size_t)b * GH;
  float ig = sigf(g[j]);
  float fg = sigf(g[H + j]);
  float gg = tanhf(g[2 * H + j]);
  float og = sigf(g[3 * H + j]);
  float cn = fg * c[idx] + ig * gg;
  c[idx] = cn;
  h[idx] = og * tanhf(cn);
}

// Fused Bahdanau attention: one block per batch row.
// scores -> softmax over S -> context. enc_proj / enc stay L2-resident.
__global__ __launch_bounds__(256) void attention_kernel(
    const float* __restrict__ enc_proj, const float* __restrict__ enc,
    const float* __restrict__ dec, const float* __restrict__ W_as,
    const float* __restrict__ b_as, float* __restrict__ ctx) {
  const int b = blockIdx.x;
  const int tid = threadIdx.x;
  __shared__ float sc[S];
  __shared__ float red[S];

  if (tid < S) {
    float acc = b_as[0];
    const float* ep = enc_proj + ((size_t)b * S + tid) * ADIM;
    const float* dc = dec + (size_t)b * ADIM;
    for (int a = 0; a < ADIM; ++a) acc += W_as[a] * tanhf(ep[a] + dc[a]);
    sc[tid] = acc;
  }
  __syncthreads();

  // softmax over S=128: max-reduce, exp, sum-reduce
  if (tid < S) red[tid] = sc[tid];
  __syncthreads();
  for (int st = S / 2; st > 0; st >>= 1) {
    if (tid < st) red[tid] = fmaxf(red[tid], red[tid + st]);
    __syncthreads();
  }
  float mx = red[0];
  __syncthreads();
  if (tid < S) {
    float e = expf(sc[tid] - mx);
    sc[tid] = e;
    red[tid] = e;
  }
  __syncthreads();
  for (int st = S / 2; st > 0; st >>= 1) {
    if (tid < st) red[tid] += red[tid + st];
    __syncthreads();
  }
  float inv = 1.0f / red[0];
  __syncthreads();
  if (tid < S) sc[tid] *= inv;
  __syncthreads();

  // ctx[b][d] = sum_s w[s] * enc[b][s][d]; coalesced across d
  for (int d = tid; d < D; d += 256) {
    float acc = 0.0f;
    for (int ss = 0; ss < S; ++ss)
      acc += sc[ss] * enc[((size_t)b * S + ss) * D + d];
    ctx[(size_t)b * D + d] = acc;
  }
}

__global__ void build_y_kernel(float* __restrict__ Y, const float* __restrict__ h,
                               const float* __restrict__ ctx) {
  int idx = blockIdx.x * blockDim.x + threadIdx.x;
  if (idx >= B * KY) return;
  int b = idx / KY, k = idx % KY;
  Y[idx] = (k < H) ? h[(size_t)b * H + k] : ctx[(size_t)b * D + (k - H)];
}

}  // namespace

// ---------------------------------------------------------------------------
// Host-side orchestration
// ---------------------------------------------------------------------------
extern "C" void kernel_launch(void* const* d_in, const int* in_sizes, int n_in,
                              void* d_out, int out_size, void* d_ws, size_t ws_size,
                              hipStream_t stream) {
  (void)in_sizes; (void)n_in; (void)out_size; (void)ws_size;
  const float* enc  = (const float*)d_in[0];
  const int*   tgt  = (const int*)d_in[1];
  const float* emb  = (const float*)d_in[2];
  const float* W_ih = (const float*)d_in[3];
  const float* b_ih = (const float*)d_in[4];
  const float* W_hh = (const float*)d_in[5];
  const float* b_hh = (const float*)d_in[6];
  const float* W_ep = (const float*)d_in[7];
  const float* b_ep = (const float*)d_in[8];
  const float* W_dp = (const float*)d_in[9];
  const float* b_dp = (const float*)d_in[10];
  const float* W_as = (const float*)d_in[11];
  const float* b_as = (const float*)d_in[12];
  const float* W_fc = (const float*)d_in[13];
  const float* b_fc = (const float*)d_in[14];
  const float* W_h0 = (const float*)d_in[15];
  const float* b_h0 = (const float*)d_in[16];
  const float* W_c0 = (const float*)d_in[17];
  const float* b_c0 = (const float*)d_in[18];
  float* out = (float*)d_out;

  // Bump allocator over workspace (256B aligned)
  char* p = (char*)d_ws;
  auto alloc = [&](size_t bytes) -> void* {
    void* r = (void*)p;
    p += (bytes + 255) & ~(size_t)255;
    return r;
  };
  float* enc_proj = (float*)alloc(sizeof(float) * (size_t)B * S * ADIM);   // 33.5 MB
  unsigned short* Wcat  = (unsigned short*)alloc(sizeof(short) * (size_t)GH * KX);
  unsigned short* Wep_b = (unsigned short*)alloc(sizeof(short) * (size_t)ADIM * D);
  unsigned short* Wdp_b = (unsigned short*)alloc(sizeof(short) * (size_t)ADIM * H);
  unsigned short* Wfc_b = (unsigned short*)alloc(sizeof(short) * (size_t)V_PAD * KY);
  unsigned short* Wh0_b = (unsigned short*)alloc(sizeof(short) * (size_t)H * D);
  unsigned short* Wc0_b = (unsigned short*)alloc(sizeof(short) * (size_t)H * D);
  float* bias_c = (float*)alloc(sizeof(float) * GH);
  float* mean_b = (float*)alloc(sizeof(float) * (size_t)B * D);
  float* h_buf  = (float*)alloc(sizeof(float) * (size_t)B * H);
  float* c_buf  = (float*)alloc(sizeof(float) * (size_t)B * H);
  float* ctx_b  = (float*)alloc(sizeof(float) * (size_t)B * D);
  float* X_buf  = (float*)alloc(sizeof(float) * (size_t)B * KX);
  float* g_buf  = (float*)alloc(sizeof(float) * (size_t)B * GH);
  float* dec_b  = (float*)alloc(sizeof(float) * (size_t)B * ADIM);
  float* Y_buf  = (float*)alloc(sizeof(float) * (size_t)B * KY);

  auto blocks = [](int n) { return (n + 255) / 256; };
  // M % 128 == 0, K % 64 == 0, Np % 64 == 0 guaranteed at every call site.
  auto gemm = [&](const float* Aa, int lda, const unsigned short* Wb, int ldw,
                  const float* bias, float* C, int ldc,
                  int M, int Np, int Ns, int K, int act) {
    dim3 g(M / TM, Np / TN);
    gemm_bf16_wmma<<<g, 256, 0, stream>>>(Aa, lda, Wb, ldw, bias, C, ldc,
                                          M, Ns, K, act);
  };

  // ---- Prep: bf16 weight images (rebuilt every call; deterministic) ----
  cvt_rows_kernel<<<blocks(GH * (E + D)), 256, 0, stream>>>(Wcat, KX, 0, W_ih, GH, E + D);
  cvt_rows_kernel<<<blocks(GH * H), 256, 0, stream>>>(Wcat, KX, E + D, W_hh, GH, H);
  cvt_rows_kernel<<<blocks(ADIM * D), 256, 0, stream>>>(Wep_b, D, 0, W_ep, ADIM, D);
  cvt_rows_kernel<<<blocks(ADIM * H), 256, 0, stream>>>(Wdp_b, H, 0, W_dp, ADIM, H);
  fill0_u16_kernel<<<blocks(V_PAD * KY), 256, 0, stream>>>(Wfc_b, V_PAD * KY);
  cvt_rows_kernel<<<blocks(V * KY), 256, 0, stream>>>(Wfc_b, KY, 0, W_fc, V, KY);
  cvt_rows_kernel<<<blocks(H * D), 256, 0, stream>>>(Wh0_b, D, 0, W_h0, H, D);
  cvt_rows_kernel<<<blocks(H * D), 256, 0, stream>>>(Wc0_b, D, 0, W_c0, H, D);
  bias_comb_kernel<<<blocks(GH), 256, 0, stream>>>(bias_c, b_ih, b_hh);

  // ---- Init state ----
  enc_mean_kernel<<<blocks(B * D), 256, 0, stream>>>(enc, mean_b);
  gemm(mean_b, D, Wh0_b, D, b_h0, h_buf, H, B, H, H, D, /*tanh*/ 1);
  gemm(mean_b, D, Wc0_b, D, b_c0, c_buf, H, B, H, H, D, /*tanh*/ 1);
  fill0_kernel<<<blocks(B * D), 256, 0, stream>>>(ctx_b, B * D);

  // ---- Encoder projection (one-time large GEMM, lands in L2) ----
  gemm(enc, D, Wep_b, D, b_ep, enc_proj, ADIM, B * S, ADIM, ADIM, D, 0);

  // ---- Serial decode recurrence ----
  for (int t = 0; t < T_STEPS; ++t) {
    build_x_kernel<<<blocks(B * KX), 256, 0, stream>>>(X_buf, emb, tgt, ctx_b, h_buf, t);
    gemm(X_buf, KX, Wcat, KX, bias_c, g_buf, GH, B, GH, GH, KX, 0);      // gates
    lstm_kernel<<<blocks(B * H), 256, 0, stream>>>(g_buf, h_buf, c_buf); // h2, c2
    gemm(h_buf, H, Wdp_b, H, b_dp, dec_b, ADIM, B, ADIM, ADIM, H, 0);    // dec proj
    attention_kernel<<<B, 256, 0, stream>>>(enc_proj, enc, dec_b, W_as, b_as, ctx_b);
    build_y_kernel<<<blocks(B * KY), 256, 0, stream>>>(Y_buf, h_buf, ctx_b);
    // logits -> d_out[b, t, :], ldc = T*V so each step writes its time slice
    gemm(Y_buf, KY, Wfc_b, KY, b_fc, out + (size_t)t * V, T_STEPS * V,
         B, V_PAD, V, KY, 0);
  }
}
